// HiveGNNPolicy_13554916786691
// MI455X (gfx1250) — matrix operand order repaired and checked by reading, not compiled, not run on hardware
//
#include <hip/hip_runtime.h>
#include <hip/hip_bf16.h>
#include <math.h>

// ---------------------------------------------------------------------------
// Problem constants (match reference)
// ---------------------------------------------------------------------------
#define NN   50000      // nodes
#define EE   800000     // edges (before self loops)
#define ETOT (NN + EE)  // edges incl. self loops
#define FIN  12
#define HIDC 128
#define BB   500
#define AA   5985
#define BN_EPS 1e-5f

// ---------------------------------------------------------------------------
// bf16 helpers (stored as uint16_t buffers; vector type only for the builtin)
// ---------------------------------------------------------------------------
typedef __bf16        v16bf  __attribute__((ext_vector_type(16)));
typedef float         v8f    __attribute__((ext_vector_type(8)));
typedef unsigned int  u32x4  __attribute__((ext_vector_type(4)));

union Frag16 { v16bf v; u32x4 q[2]; };
union FragC  { v8f   v; float f[8]; };

static __device__ __forceinline__ unsigned short f32_to_bf16u(float f) {
  unsigned int u = __float_as_uint(f);
  if ((u & 0x7fffffffu) > 0x7f800000u) return (unsigned short)((u >> 16) | 0x40); // quiet NaN
  u += 0x7fffu + ((u >> 16) & 1u);  // round to nearest even
  return (unsigned short)(u >> 16);
}

static __device__ __forceinline__ void atomicMaxF(float* addr, float val) {
  unsigned int* ua = (unsigned int*)addr;
  unsigned int old = __hip_atomic_load(ua, __ATOMIC_RELAXED, __HIP_MEMORY_SCOPE_AGENT);
  while (__uint_as_float(old) < val) {
    unsigned int assumed = old;
    old = atomicCAS(ua, assumed, __float_as_uint(val));
    if (old == assumed) break;
  }
}

// ---------------------------------------------------------------------------
// Generic bf16 WMMA GEMM:  out[M,N] = A[M,K] @ Wt[N,K]^T + bias  (f32 accum)
// A, Wt bf16 (uint16).  K compile-time (64/128/256) so the k-loop unrolls to
// back-to-back v_wmma with clause-batched b128 fragment loads.
//
// No predication in the hot loop: out-of-range A rows / B cols are *clamped*
// to a valid address (no OOB) — they only feed C rows/cols that the guarded
// epilogue never stores.
//
// Block = 128 threads = 4 waves; each wave owns one 16x16 C tile.
// Fragment layouts per CDNA5 ISA 7.12.2 (wave32):
//   A (16-bit 16x32): lane<16 -> row=lane, K 0..7 & 16..23; lane>=16 -> +8
//     => per lane two contiguous 16B runs at +akh and +16+akh
//   B (16-bit 32x16): lane<16 -> col=lane, K 0..7 & 16..23; lane>=16 -> +8
//     => per lane two contiguous 16B runs at +bkh and +16+bkh
//   C (f32 16x16): VGPR r -> (r + (lane>=16)*8, lane&15)
// ---------------------------------------------------------------------------
template <int K>
__global__ void wmma_gemm_bf16(const unsigned short* __restrict__ A,
                               const unsigned short* __restrict__ Wt,
                               const float* __restrict__ bias,
                               float* __restrict__ out,
                               unsigned short* __restrict__ out_bf,
                               const int* __restrict__ mask,
                               int M, int N, int act) {
  const int lane  = threadIdx.x & 31;
  const int wave  = threadIdx.x >> 5;
  const int ntile = blockIdx.x * 4 + wave;
  const int n0    = ntile * 16;
  const int row0  = blockIdx.y * 16;
  if (n0 >= N) return;                       // wave-uniform exit

  FragC acc;
#pragma unroll
  for (int i = 0; i < 8; ++i) acc.f[i] = 0.f;

  const int akh = (lane >> 4) << 3;          // 0 or 8
  const int bkh = (lane >> 4) << 3;          // 0 or 8
  int arow = row0 + (lane & 15); if (arow >= M) arow = M - 1;  // clamp, no branch in loop
  int bcol = n0   + (lane & 15); if (bcol >= N) bcol = N - 1;
  const unsigned short* Ap = A  + (size_t)arow * K + akh;
  const unsigned short* Bp = Wt + (size_t)bcol * K + bkh;

#pragma unroll
  for (int kb = 0; kb < K; kb += 32) {
    Frag16 a, b;
    a.q[0] = *(const u32x4*)(Ap + kb);        // K = kb+akh+ 0.. 7
    a.q[1] = *(const u32x4*)(Ap + kb + 16);   // K = kb+akh+16..23
    b.q[0] = *(const u32x4*)(Bp + kb);        // K = kb+bkh+ 0.. 7
    b.q[1] = *(const u32x4*)(Bp + kb + 16);   // K = kb+bkh+16..23
    acc.v = __builtin_amdgcn_wmma_f32_16x16x32_bf16(false, a.v, false, b.v,
                                                    (short)0, acc.v, false, false);
  }

  const int cn    = n0 + (lane & 15);
  const int rbase = row0 + ((lane >> 4) << 3);
  if (cn < N) {
    const float bv = bias ? bias[cn] : 0.f;
#pragma unroll
    for (int r = 0; r < 8; ++r) {
      const int row = rbase + r;
      if (row < M) {
        float v = acc.f[r] + bv;
        if (act) v = v > 0.f ? v : 0.f;
        if (mask) v = mask[(size_t)row * N + cn] ? v : -INFINITY;
        out[(size_t)row * N + cn] = v;
        if (out_bf) out_bf[(size_t)row * N + cn] = f32_to_bf16u(v);
      }
    }
  }
}

// ---------------------------------------------------------------------------
// Weight pack: W[K,N] fp32 -> Wt[N,K] bf16 (transpose + convert)
// ---------------------------------------------------------------------------
__global__ void pack_wt_kernel(const float* __restrict__ W,
                               unsigned short* __restrict__ Wt, int K, int N) {
  const int idx = blockIdx.x * blockDim.x + threadIdx.x;
  if (idx >= K * N) return;
  const int n = idx / K, k = idx - n * K;
  Wt[idx] = f32_to_bf16u(W[(size_t)k * N + n]);
}

__global__ void fill_f32_kernel(float* __restrict__ p, float v, int n) {
  const int i = blockIdx.x * blockDim.x + threadIdx.x;
  if (i < n) p[i] = v;
}

// ---------------------------------------------------------------------------
// Embedding: h = relu(x @ emb_W + emb_b), also emit bf16 copy
// ---------------------------------------------------------------------------
__global__ void embed_kernel(const float* __restrict__ x,
                             const float* __restrict__ W,
                             const float* __restrict__ b,
                             float* __restrict__ h,
                             unsigned short* __restrict__ hbf) {
  const int idx = blockIdx.x * blockDim.x + threadIdx.x;
  if (idx >= NN * HIDC) return;
  const int n = idx >> 7, c = idx & 127;
  float acc = b[c];
#pragma unroll
  for (int k = 0; k < FIN; ++k) acc += x[n * FIN + k] * W[k * HIDC + c];
  acc = acc > 0.f ? acc : 0.f;
  h[idx] = acc;
  hbf[idx] = f32_to_bf16u(acc);
}

// ---------------------------------------------------------------------------
// Edge pass 1: logit[e,h] = att[h] . leaky(xl[src]+xr[dst]); atomic seg-max.
// One wave per edge; lane = channel-within-head (coalesced across lanes).
// ---------------------------------------------------------------------------
__global__ void edge_logit_kernel(const float* __restrict__ xl,
                                  const float* __restrict__ xr,
                                  const int* __restrict__ eidx,
                                  const float* __restrict__ att,
                                  float* __restrict__ logit,
                                  float* __restrict__ mmax) {
  const int wid  = (blockIdx.x * blockDim.x + threadIdx.x) >> 5;
  const int lane = threadIdx.x & 31;
  if (wid >= ETOT) return;
  const int s = wid < EE ? eidx[wid]      : (wid - EE);   // src (self loop tail)
  const int d = wid < EE ? eidx[EE + wid] : (wid - EE);   // dst
  float p[4];
#pragma unroll
  for (int hh = 0; hh < 4; ++hh) {
    const int c = hh * 32 + lane;
    float v = xl[(size_t)s * HIDC + c] + xr[(size_t)d * HIDC + c];
    v = v > 0.f ? v : 0.2f * v;                            // LeakyReLU(0.2)
    p[hh] = v * att[c];
  }
#pragma unroll
  for (int off = 16; off > 0; off >>= 1) {
#pragma unroll
    for (int hh = 0; hh < 4; ++hh) p[hh] += __shfl_xor(p[hh], off, 32);
  }
  if (lane == 0) {
#pragma unroll
    for (int hh = 0; hh < 4; ++hh) {
      logit[(size_t)wid * 4 + hh] = p[hh];
      atomicMaxF(&mmax[(size_t)d * 4 + hh], p[hh]);
    }
  }
}

// Edge pass 2: w = exp(logit - max[dst]); atomic seg-sum.  Thread per (e,h).
__global__ void edge_exp_kernel(const float* __restrict__ logit,
                                const float* __restrict__ mmax,
                                const int* __restrict__ eidx,
                                float* __restrict__ wbuf,
                                float* __restrict__ denom) {
  const int idx = blockIdx.x * blockDim.x + threadIdx.x;
  if (idx >= ETOT * 4) return;
  const int e = idx >> 2, hh = idx & 3;
  const int d = e < EE ? eidx[EE + e] : (e - EE);
  const float wv = __expf(logit[idx] - mmax[(size_t)d * 4 + hh]);
  wbuf[idx] = wv;
  atomicAdd(&denom[(size_t)d * 4 + hh], wv);
}

// Edge pass 3: agg[dst] += (w/denom[dst]) * xl[src].  Wave per edge.
__global__ void edge_agg_kernel(const float* __restrict__ xl,
                                const float* __restrict__ wbuf,
                                const float* __restrict__ denom,
                                const int* __restrict__ eidx,
                                float* __restrict__ agg) {
  const int wid  = (blockIdx.x * blockDim.x + threadIdx.x) >> 5;
  const int lane = threadIdx.x & 31;
  if (wid >= ETOT) return;
  const int s = wid < EE ? eidx[wid]      : (wid - EE);
  const int d = wid < EE ? eidx[EE + wid] : (wid - EE);
  float al[4];
#pragma unroll
  for (int hh = 0; hh < 4; ++hh)
    al[hh] = wbuf[(size_t)wid * 4 + hh] / denom[(size_t)d * 4 + hh];
#pragma unroll
  for (int hh = 0; hh < 4; ++hh) {
    const int c = hh * 32 + lane;
    atomicAdd(&agg[(size_t)d * HIDC + c], al[hh] * xl[(size_t)s * HIDC + c]);
  }
}

// ---------------------------------------------------------------------------
// BatchNorm (training-mode batch stats, biased var) + ReLU + residual.
// Note: p["bias"] (pre-BN additive bias) cancels exactly under BN -> skipped.
// ---------------------------------------------------------------------------
__global__ void bn_stats_kernel(const float* __restrict__ a,
                                float* __restrict__ sum,
                                float* __restrict__ sumsq) {
  const int c = threadIdx.x;  // 128 channels
  float s = 0.f, ss = 0.f;
  for (int r = blockIdx.x; r < NN; r += gridDim.x) {
    const float v = a[(size_t)r * HIDC + c];
    s += v; ss += v * v;
  }
  atomicAdd(&sum[c], s);
  atomicAdd(&sumsq[c], ss);
}

__global__ void bn_apply_kernel(const float* __restrict__ a,
                                const float* __restrict__ sum,
                                const float* __restrict__ sumsq,
                                const float* __restrict__ gamma,
                                const float* __restrict__ beta,
                                float* __restrict__ h,
                                unsigned short* __restrict__ hbf, int resflag) {
  const int idx = blockIdx.x * blockDim.x + threadIdx.x;
  if (idx >= NN * HIDC) return;
  const int c = idx & 127;
  const float invn = 1.0f / (float)NN;
  const float mu  = sum[c] * invn;
  const float var = sumsq[c] * invn - mu * mu;
  float y = (a[idx] - mu) * rsqrtf(var + BN_EPS) * gamma[c] + beta[c];
  y = y > 0.f ? y : 0.f;
  const float hn = y + (resflag ? h[idx] : 0.f);
  h[idx] = hn;
  hbf[idx] = f32_to_bf16u(hn);
}

// ---------------------------------------------------------------------------
// Global mean+max pooling per graph (uses `batch` segment ids)
// ---------------------------------------------------------------------------
__global__ void pool_count_kernel(const int* __restrict__ batch,
                                  float* __restrict__ counts) {
  const int n = blockIdx.x * blockDim.x + threadIdx.x;
  if (n < NN) atomicAdd(&counts[batch[n]], 1.0f);
}

__global__ void pool_accum_kernel(const float* __restrict__ h,
                                  const int* __restrict__ batch,
                                  float* __restrict__ gsum,
                                  float* __restrict__ gmax) {
  const int idx = blockIdx.x * blockDim.x + threadIdx.x;
  if (idx >= NN * HIDC) return;
  const int n = idx >> 7, c = idx & 127;
  const int b = batch[n];
  const float v = h[idx];
  atomicAdd(&gsum[(size_t)b * HIDC + c], v);
  atomicMaxF(&gmax[(size_t)b * HIDC + c], v);
}

__global__ void pool_final_kernel(const float* __restrict__ gsum,
                                  const float* __restrict__ gmax,
                                  const float* __restrict__ counts,
                                  float* __restrict__ g,
                                  unsigned short* __restrict__ gbf) {
  const int idx = blockIdx.x * blockDim.x + threadIdx.x;
  if (idx >= BB * 2 * HIDC) return;
  const int b = idx >> 8, c = idx & 255;
  const float v = (c < HIDC) ? gsum[(size_t)b * HIDC + c] / counts[b]
                             : gmax[(size_t)b * HIDC + (c - HIDC)];
  g[idx] = v;
  gbf[idx] = f32_to_bf16u(v);
}

// Value head tail: [500,64] @ [64,1] + b, tanh
__global__ void value_final_kernel(const float* __restrict__ vh,
                                   const float* __restrict__ W,
                                   const float* __restrict__ b,
                                   float* __restrict__ out) {
  const int r = blockIdx.x * blockDim.x + threadIdx.x;
  if (r >= BB) return;
  float acc = b[0];
#pragma unroll
  for (int k = 0; k < 64; ++k) acc += vh[r * 64 + k] * W[k];
  out[r] = tanhf(acc);
}

// ---------------------------------------------------------------------------
// Host launcher
// ---------------------------------------------------------------------------
extern "C" void kernel_launch(void* const* d_in, const int* in_sizes, int n_in,
                              void* d_out, int out_size, void* d_ws, size_t ws_size,
                              hipStream_t stream) {
  (void)in_sizes; (void)n_in; (void)out_size; (void)ws_size;

  // Input order: setup_inputs() dict order; nested `params` flattened as
  // jax tree leaves (dict keys sorted, lists in order):
  //  0:x  1:emb_W 2:emb_b
  //  3+8i: gat[i] = {Wl, Wr, att, beta, bias(unused), bl, br, gamma}
  //  35..40: policy (W1,b1,W2,b2,W3,b3)   41..46: value (W1,b1,W2,b2,W3,b3)
  //  47: edge_index  48: batch  49: action_mask
  const float* x    = (const float*)d_in[0];
  const float* embW = (const float*)d_in[1];
  const float* embB = (const float*)d_in[2];
  const float *Wl[4], *Wr[4], *attp[4], *betap[4], *blp[4], *brp[4], *gammap[4];
  for (int i = 0; i < 4; ++i) {
    const int base = 3 + 8 * i;
    Wl[i]     = (const float*)d_in[base + 0];
    Wr[i]     = (const float*)d_in[base + 1];
    attp[i]   = (const float*)d_in[base + 2];
    betap[i]  = (const float*)d_in[base + 3];
    /* d_in[base+4] = bias: no-op under immediately-following batch-stat BN */
    blp[i]    = (const float*)d_in[base + 5];
    brp[i]    = (const float*)d_in[base + 6];
    gammap[i] = (const float*)d_in[base + 7];
  }
  const float* pW1 = (const float*)d_in[35]; const float* pb1 = (const float*)d_in[36];
  const float* pW2 = (const float*)d_in[37]; const float* pb2 = (const float*)d_in[38];
  const float* pW3 = (const float*)d_in[39]; const float* pb3 = (const float*)d_in[40];
  const float* vW1 = (const float*)d_in[41]; const float* vb1 = (const float*)d_in[42];
  const float* vW2 = (const float*)d_in[43]; const float* vb2 = (const float*)d_in[44];
  const float* vW3 = (const float*)d_in[45]; const float* vb3 = (const float*)d_in[46];
  const int* eidx  = (const int*)d_in[47];
  const int* batch = (const int*)d_in[48];
  const int* amask = (const int*)d_in[49];
  float* outp = (float*)d_out;  // [500*5985] policy logits, then [500] value

  // -------- workspace carve-up (256B aligned) --------
  char* wp = (char*)d_ws;
  auto carve = [&](size_t bytes) {
    char* p = wp; wp += (bytes + 255) & ~(size_t)255; return (void*)p;
  };
  float*          h     = (float*)carve((size_t)NN * HIDC * 4);
  unsigned short* hbf   = (unsigned short*)carve((size_t)NN * HIDC * 2);
  float*          xl    = (float*)carve((size_t)NN * HIDC * 4);
  float*          xr    = (float*)carve((size_t)NN * HIDC * 4);
  float*          agg   = (float*)carve((size_t)NN * HIDC * 4);
  float*          logit = (float*)carve((size_t)ETOT * 4 * 4);
  float*          wbuf  = (float*)carve((size_t)ETOT * 4 * 4);
  float*          mmax  = (float*)carve((size_t)NN * 4 * 4);
  float*          denom = (float*)carve((size_t)NN * 4 * 4);
  float*          bnsum = (float*)carve(HIDC * 4);
  float*          bnss  = (float*)carve(HIDC * 4);
  unsigned short* WlT[4]; unsigned short* WrT[4];
  for (int i = 0; i < 4; ++i) {
    WlT[i] = (unsigned short*)carve(HIDC * HIDC * 2);
    WrT[i] = (unsigned short*)carve(HIDC * HIDC * 2);
  }
  unsigned short* pW1T = (unsigned short*)carve(256 * 128 * 2);
  unsigned short* pW2T = (unsigned short*)carve(128 * 128 * 2);
  unsigned short* pW3T = (unsigned short*)carve((size_t)128 * AA * 2);
  unsigned short* vW1T = (unsigned short*)carve(256 * 128 * 2);
  unsigned short* vW2T = (unsigned short*)carve(128 * 64 * 2);
  float*          gsum  = (float*)carve((size_t)BB * HIDC * 4);
  float*          gmax  = (float*)carve((size_t)BB * HIDC * 4);
  float*          cnts  = (float*)carve(BB * 4);
  float*          g     = (float*)carve((size_t)BB * 256 * 4);
  unsigned short* gbf   = (unsigned short*)carve((size_t)BB * 256 * 2);
  float*          ph1   = (float*)carve((size_t)BB * 128 * 4);
  unsigned short* ph1b  = (unsigned short*)carve((size_t)BB * 128 * 2);
  float*          ph2   = (float*)carve((size_t)BB * 128 * 4);
  unsigned short* ph2b  = (unsigned short*)carve((size_t)BB * 128 * 2);
  float*          vh1   = (float*)carve((size_t)BB * 128 * 4);
  unsigned short* vh1b  = (unsigned short*)carve((size_t)BB * 128 * 2);
  float*          vh2   = (float*)carve((size_t)BB * 64 * 4);
  unsigned short* vh2b  = (unsigned short*)carve((size_t)BB * 64 * 2);
  (void)vh2b;

  auto fill = [&](float* p, float v, int n) {
    fill_f32_kernel<<<(n + 255) / 256, 256, 0, stream>>>(p, v, n);
  };
  auto pack = [&](const float* W, unsigned short* Wt, int K, int Nc) {
    pack_wt_kernel<<<(K * Nc + 255) / 256, 256, 0, stream>>>(W, Wt, K, Nc);
  };
  auto gemm = [&](const unsigned short* A, const unsigned short* Wt,
                  const float* bias, float* out, unsigned short* obf,
                  const int* mk, int M, int Nc, int K, int act) {
    const int ntiles = (Nc + 15) / 16;
    dim3 grid((ntiles + 3) / 4, (M + 15) / 16);
    if (K == 64)
      wmma_gemm_bf16<64><<<grid, dim3(128), 0, stream>>>(A, Wt, bias, out, obf, mk, M, Nc, act);
    else if (K == 128)
      wmma_gemm_bf16<128><<<grid, dim3(128), 0, stream>>>(A, Wt, bias, out, obf, mk, M, Nc, act);
    else
      wmma_gemm_bf16<256><<<grid, dim3(128), 0, stream>>>(A, Wt, bias, out, obf, mk, M, Nc, act);
  };

  // -------- pack weights (bf16, transposed) --------
  for (int i = 0; i < 4; ++i) { pack(Wl[i], WlT[i], HIDC, HIDC); pack(Wr[i], WrT[i], HIDC, HIDC); }
  pack(pW1, pW1T, 256, 128); pack(pW2, pW2T, 128, 128); pack(pW3, pW3T, 128, AA);
  pack(vW1, vW1T, 256, 128); pack(vW2, vW2T, 128, 64);

  // -------- embedding --------
  embed_kernel<<<(NN * HIDC + 255) / 256, 256, 0, stream>>>(x, embW, embB, h, hbf);

  // -------- GAT layers --------
  const int edgeWaveBlocks = (ETOT * 32 + 255) / 256;
  for (int i = 0; i < 4; ++i) {
    gemm(hbf, WlT[i], blp[i], xl, nullptr, nullptr, NN, HIDC, HIDC, 0);
    gemm(hbf, WrT[i], brp[i], xr, nullptr, nullptr, NN, HIDC, HIDC, 0);
    fill(mmax, -INFINITY, NN * 4);
    fill(denom, 0.f, NN * 4);
    fill(agg, 0.f, NN * HIDC);
    edge_logit_kernel<<<edgeWaveBlocks, 256, 0, stream>>>(xl, xr, eidx, attp[i], logit, mmax);
    edge_exp_kernel<<<(ETOT * 4 + 255) / 256, 256, 0, stream>>>(logit, mmax, eidx, wbuf, denom);
    edge_agg_kernel<<<edgeWaveBlocks, 256, 0, stream>>>(xl, wbuf, denom, eidx, agg);
    fill(bnsum, 0.f, HIDC);
    fill(bnss, 0.f, HIDC);
    bn_stats_kernel<<<512, HIDC, 0, stream>>>(agg, bnsum, bnss);
    bn_apply_kernel<<<(NN * HIDC + 255) / 256, 256, 0, stream>>>(
        agg, bnsum, bnss, gammap[i], betap[i], h, hbf, i > 0 ? 1 : 0);
  }

  // -------- pooling --------
  fill(gsum, 0.f, BB * HIDC);
  fill(gmax, -INFINITY, BB * HIDC);
  fill(cnts, 0.f, BB);
  pool_count_kernel<<<(NN + 255) / 256, 256, 0, stream>>>(batch, cnts);
  pool_accum_kernel<<<(NN * HIDC + 255) / 256, 256, 0, stream>>>(h, batch, gsum, gmax);
  pool_final_kernel<<<(BB * 256 + 255) / 256, 256, 0, stream>>>(gsum, gmax, cnts, g, gbf);

  // -------- policy head (mask fused into final GEMM store) --------
  gemm(gbf,  pW1T, pb1, ph1, ph1b, nullptr, BB, 128, 256, 1);
  gemm(ph1b, pW2T, pb2, ph2, ph2b, nullptr, BB, 128, 128, 1);
  gemm(ph2b, pW3T, pb3, outp, nullptr, amask, BB, AA, 128, 0);

  // -------- value head --------
  gemm(gbf,  vW1T, vb1, vh1, vh1b, nullptr, BB, 128, 256, 1);
  gemm(vh1b, vW2T, vb2, vh2, nullptr, nullptr, BB, 64, 128, 1);
  value_final_kernel<<<(BB + 255) / 256, 256, 0, stream>>>(
      vh2, vW3, vb3, outp + (size_t)BB * AA);
}
